// Swin_19284403159608
// MI455X (gfx1250) — matrix operand order repaired and checked by reading, not compiled, not run on hardware
//
#include <hip/hip_runtime.h>
#include <math.h>

// ---------------------------------------------------------------------------
// Swin-T forward, MI455X (gfx1250, wave32, WMMA 16x16x32 bf16).
// bf16 end-to-end on the matmul path: LayerNorm / GELU / attention emit bf16,
// weights are pre-converted to bf16 in workspace once per launch. The fp32
// residual trunk (x) stays fp32. GEMM stages the A tile in LDS (bf16, shared
// by all 4 waves); B tiles load directly in WMMA B-layout from the bf16
// weight table. Async global->LDS (ASYNCcnt) is used for A staging when the
// toolchain exposes the builtin.
// ---------------------------------------------------------------------------

typedef unsigned short u16;
typedef __attribute__((ext_vector_type(16))) __bf16 v16bf;
typedef __attribute__((ext_vector_type(8)))  float  v8f;
typedef __attribute__((ext_vector_type(4)))  int    v4i;

#if defined(__has_builtin)
#if __has_builtin(__builtin_amdgcn_global_load_async_to_lds_b128) && \
    __has_builtin(__builtin_amdgcn_s_wait_asynccnt)
#define HAVE_ASYNC 1
#endif
#endif

#ifdef HAVE_ASYNC
typedef __attribute__((address_space(1))) v4i gv4i;
typedef __attribute__((address_space(3))) v4i lv4i;
#endif

union V16 {
    v16bf v;
    __bf16 h[16];
    u16 u[16];
    uint4 q[2];
};
union V8 {
    v8f v;
    float f[8];
};

__device__ inline u16 bfbits(float f) {
    union { __bf16 b; u16 u; } c;
    c.b = (__bf16)f;
    return c.u;
}

// ---------------------------------------------------------------------------
// GEMM: out[M,N] = A[M,K](bf16) @ W[N,K](bf16)^T (+bias) (+GELU) (+fp32 res)
// Block = 128 threads (4 waves), tile 64(M) x 128(N); each wave owns two
// 16-col subtiles. A tile (64x32 bf16) staged in LDS per K-step.
// K is a multiple of 32 at every call site.
// ---------------------------------------------------------------------------
__global__ __launch_bounds__(128)
void gemm_bf16_wmma(const u16* __restrict__ A, const u16* __restrict__ Wb,
                    const float* __restrict__ bias, const float* __restrict__ res,
                    void* __restrict__ outp, int M, int N, int K,
                    int gelu, int out_bf16)
{
    __shared__ __align__(16) u16 Asm[64 * 32];

    const int tid  = threadIdx.x;
    const int lane = tid & 31, wv = tid >> 5;
    const int half = lane >> 4, l16 = lane & 15;
    const int m0 = blockIdx.y * 64;
    const int n0 = blockIdx.x * 128 + wv * 32;

    V8 acc[4][2];
#pragma unroll
    for (int mt = 0; mt < 4; ++mt)
#pragma unroll
        for (int nt = 0; nt < 2; ++nt)
#pragma unroll
            for (int j = 0; j < 8; ++j) acc[mt][nt].f[j] = 0.0f;

    int nr0 = n0 + l16;      if (nr0 >= N) nr0 = N - 1;
    int nr1 = n0 + 16 + l16; if (nr1 >= N) nr1 = N - 1;
    const u16* wrow0 = Wb + (size_t)nr0 * K + 16 * half;
    const u16* wrow1 = Wb + (size_t)nr1 * K + 16 * half;

    // A staging: thread copies 32B (16 bf16): row = tid>>1, col = (tid&1)*16
    const int arow = tid >> 1;
    const int acol = (tid & 1) * 16;
    int mrow = m0 + arow; if (mrow >= M) mrow = M - 1;
    const u16* agp = A + (size_t)mrow * K + acol;
    u16* lds_dst = &Asm[arow * 32 + acol];

#ifndef HAVE_ASYNC
    uint4 pa0 = *(const uint4*)(agp);
    uint4 pa1 = *(const uint4*)(agp + 8);
#endif

    for (int k0 = 0; k0 < K; k0 += 32) {
        V16 bv0, bv1;
        {
            const uint4* wp0 = (const uint4*)(wrow0 + k0);
            const uint4* wp1 = (const uint4*)(wrow1 + k0);
            bv0.q[0] = wp0[0]; bv0.q[1] = wp0[1];
            bv1.q[0] = wp1[0]; bv1.q[1] = wp1[1];
        }

        __syncthreads();   // previous tile fully consumed
#ifdef HAVE_ASYNC
        __builtin_amdgcn_global_load_async_to_lds_b128(
            (gv4i*)(agp + k0), (lv4i*)lds_dst, 0, 0);
        __builtin_amdgcn_global_load_async_to_lds_b128(
            (gv4i*)(agp + k0 + 8), (lv4i*)(lds_dst + 8), 0, 0);
        __builtin_amdgcn_s_wait_asynccnt(0);
#else
        *(uint4*)lds_dst = pa0;
        *(uint4*)(lds_dst + 8) = pa1;
        if (k0 + 32 < K) {   // software pipeline: prefetch next A chunk
            pa0 = *(const uint4*)(agp + k0 + 32);
            pa1 = *(const uint4*)(agp + k0 + 40);
        }
#endif
        __syncthreads();

#pragma unroll
        for (int mt = 0; mt < 4; ++mt) {
            V16 av;
            av.q[0] = *(const uint4*)&Asm[(mt * 16 + l16) * 32 + 16 * half];
            av.q[1] = *(const uint4*)&Asm[(mt * 16 + l16) * 32 + 16 * half + 8];
            acc[mt][0].v = __builtin_amdgcn_wmma_f32_16x16x32_bf16(
                false, av.v, false, bv0.v, (short)0, acc[mt][0].v, false, false);
            acc[mt][1].v = __builtin_amdgcn_wmma_f32_16x16x32_bf16(
                false, av.v, false, bv1.v, (short)0, acc[mt][1].v, false, false);
        }
    }

#pragma unroll
    for (int nt = 0; nt < 2; ++nt) {
        const int n = n0 + nt * 16 + l16;
        if (n < N) {
            const float bsum = bias ? bias[n] : 0.0f;
#pragma unroll
            for (int mt = 0; mt < 4; ++mt) {
#pragma unroll
                for (int j = 0; j < 8; ++j) {
                    int m = m0 + mt * 16 + half * 8 + j;
                    if (m < M) {
                        float v = acc[mt][nt].f[j] + bsum;
                        if (gelu) v = 0.5f * v * (1.0f + erff(v * 0.70710678f));
                        if (res) v += res[(size_t)m * N + n];
                        if (out_bf16) ((u16*)outp)[(size_t)m * N + n] = bfbits(v);
                        else ((float*)outp)[(size_t)m * N + n] = v;
                    }
                }
            }
        }
    }
}

// ---------------------------------------------------------------------------
// Windowed attention, one wave per (batch, window, head); head_dim = 32.
// qkv input and output are bf16; 1/sqrt(32) scale folded into the logits.
// ---------------------------------------------------------------------------
__global__ __launch_bounds__(32)
void win_attn(const u16* __restrict__ qkv, const float* __restrict__ rpb,
              u16* __restrict__ aout, int HW, int nH, int C, int shift)
{
    const int L   = HW * HW;
    const int nWw = HW / 7;
    const int nW  = nWw * nWw;
    int id = blockIdx.x;
    const int head = id % nH; id /= nH;
    const int win  = id % nW; id /= nW;
    const int b    = id;
    const int wy = win / nWw, wx = win % nWw;
    const int lane = threadIdx.x;
    const int half = lane >> 4, l16 = lane & 15;
    const int C3 = 3 * C;

    __shared__ __align__(16) u16 Qs[64 * 32];
    __shared__ __align__(16) u16 Ks[64 * 32];
    __shared__ __align__(16) u16 Vt[32 * 64];
    __shared__ __align__(16) u16 As[64 * 64];

    for (int r = 0; r < 64; ++r) {
        u16 q = 0, k = 0, v = 0;
        if (r < 49) {
            int rr = r / 7, rc = r % 7;
            int y = wy * 7 + rr + shift; if (y >= HW) y -= HW;
            int x = wx * 7 + rc + shift; if (x >= HW) x -= HW;
            size_t base = ((size_t)b * L + y * HW + x) * C3 + head * 32 + lane;
            q = qkv[base];
            k = qkv[base + C];
            v = qkv[base + 2 * C];
        }
        Qs[r * 32 + lane] = q;
        Ks[r * 32 + lane] = k;
        Vt[lane * 64 + r] = v;
    }
    __syncthreads();

    // ---- Q @ K^T -> 4x4 tiles ---------------------------------------------
    V16 qa[4];
#pragma unroll
    for (int mt = 0; mt < 4; ++mt) {
        qa[mt].q[0] = *(const uint4*)&Qs[(mt * 16 + l16) * 32 + 16 * half];
        qa[mt].q[1] = *(const uint4*)&Qs[(mt * 16 + l16) * 32 + 16 * half + 8];
    }
    V8 sc[4][4];
#pragma unroll
    for (int mt = 0; mt < 4; ++mt)
#pragma unroll
        for (int nt = 0; nt < 4; ++nt)
#pragma unroll
            for (int j = 0; j < 8; ++j) sc[mt][nt].f[j] = 0.0f;

#pragma unroll
    for (int nt = 0; nt < 4; ++nt) {
        V16 kb;
        kb.q[0] = *(const uint4*)&Ks[(nt * 16 + l16) * 32 + 16 * half];
        kb.q[1] = *(const uint4*)&Ks[(nt * 16 + l16) * 32 + 16 * half + 8];
#pragma unroll
        for (int mt = 0; mt < 4; ++mt)
            sc[mt][nt].v = __builtin_amdgcn_wmma_f32_16x16x32_bf16(
                false, qa[mt].v, false, kb.v, (short)0, sc[mt][nt].v, false, false);
    }

    // ---- scale + bias + mask + softmax ------------------------------------
    const float scale = 0.17677669529663689f; // 1/sqrt(32)
#pragma unroll
    for (int mt = 0; mt < 4; ++mt) {
#pragma unroll
        for (int j = 0; j < 8; ++j) {
            const int i = mt * 16 + 8 * half + j;
            const int ri = i / 7, ci = i % 7;
            float s[4];
#pragma unroll
            for (int nt = 0; nt < 4; ++nt) {
                const int col = nt * 16 + l16;
                float val = -1e30f;
                if (i < 49 && col < 49) {
                    int rj = col / 7, cj = col % 7;
                    int ridx = (ri - rj + 6) * 13 + (ci - cj + 6);
                    val = sc[mt][nt].f[j] * scale + rpb[ridx * nH + head];
                    if (shift > 0) {
                        int yi = wy * 7 + ri, xi = wx * 7 + ci;
                        int yj = wy * 7 + rj, xj = wx * 7 + cj;
                        int regi = (yi < HW - 7 ? 0 : (yi < HW - shift ? 1 : 2)) * 3
                                 + (xi < HW - 7 ? 0 : (xi < HW - shift ? 1 : 2));
                        int regj = (yj < HW - 7 ? 0 : (yj < HW - shift ? 1 : 2)) * 3
                                 + (xj < HW - 7 ? 0 : (xj < HW - shift ? 1 : 2));
                        if (regi != regj) val -= 100.0f;
                    }
                }
                s[nt] = val;
            }
            float mx = fmaxf(fmaxf(s[0], s[1]), fmaxf(s[2], s[3]));
#pragma unroll
            for (int d = 1; d < 16; d <<= 1) mx = fmaxf(mx, __shfl_xor(mx, d, 32));
            float e[4], sum = 0.f;
#pragma unroll
            for (int nt = 0; nt < 4; ++nt) { e[nt] = __expf(s[nt] - mx); sum += e[nt]; }
#pragma unroll
            for (int d = 1; d < 16; d <<= 1) sum += __shfl_xor(sum, d, 32);
            const float inv = 1.0f / sum;
#pragma unroll
            for (int nt = 0; nt < 4; ++nt)
                As[i * 64 + nt * 16 + l16] = bfbits(e[nt] * inv);
        }
    }
    __syncthreads();

    // ---- attn @ V ----------------------------------------------------------
#pragma unroll
    for (int mt = 0; mt < 4; ++mt) {
        V16 aa[2];
#pragma unroll
        for (int kt = 0; kt < 2; ++kt) {
            aa[kt].q[0] = *(const uint4*)&As[(mt * 16 + l16) * 64 + kt * 32 + 16 * half];
            aa[kt].q[1] = *(const uint4*)&As[(mt * 16 + l16) * 64 + kt * 32 + 16 * half + 8];
        }
#pragma unroll
        for (int nt = 0; nt < 2; ++nt) {
            V8 o;
#pragma unroll
            for (int j = 0; j < 8; ++j) o.f[j] = 0.0f;
#pragma unroll
            for (int kt = 0; kt < 2; ++kt) {
                V16 vb;
                vb.q[0] = *(const uint4*)&Vt[(nt * 16 + l16) * 64 + kt * 32 + 16 * half];
                vb.q[1] = *(const uint4*)&Vt[(nt * 16 + l16) * 64 + kt * 32 + 16 * half + 8];
                o.v = __builtin_amdgcn_wmma_f32_16x16x32_bf16(
                    false, aa[kt].v, false, vb.v, (short)0, o.v, false, false);
            }
#pragma unroll
            for (int j = 0; j < 8; ++j) {
                int i = mt * 16 + 8 * half + j;
                if (i < 49) {
                    int rr = i / 7, rc = i % 7;
                    int y = wy * 7 + rr + shift; if (y >= HW) y -= HW;
                    int x = wx * 7 + rc + shift; if (x >= HW) x -= HW;
                    aout[((size_t)b * L + y * HW + x) * C + head * 32 + nt * 16 + l16] =
                        bfbits(o.f[j]);
                }
            }
        }
    }
}

// ---------------------------------------------------------------------------
// LayerNorm: fp32 in, bf16 out. One wave per row.
// ---------------------------------------------------------------------------
__global__ __launch_bounds__(128)
void layernorm_k(const float* __restrict__ x, const float* __restrict__ g,
                 const float* __restrict__ bta, u16* __restrict__ y,
                 int rows, int C)
{
    int row = blockIdx.x * 4 + (threadIdx.x >> 5);
    if (row >= rows) return;
    int lane = threadIdx.x & 31;
    const float* xr = x + (size_t)row * C;
    float s = 0.f, s2 = 0.f;
    for (int c = lane; c < C; c += 32) { float v = xr[c]; s += v; s2 += v * v; }
#pragma unroll
    for (int d = 1; d < 32; d <<= 1) { s += __shfl_xor(s, d, 32); s2 += __shfl_xor(s2, d, 32); }
    float mu = s / C;
    float rstd = rsqrtf(s2 / C - mu * mu + 1e-5f);
    u16* yr = y + (size_t)row * C;
    for (int c = lane; c < C; c += 32) yr[c] = bfbits((xr[c] - mu) * rstd * g[c] + bta[c]);
}

// ---------------------------------------------------------------------------
// Patch embed: 4x4 s4 conv (3->96) + bias + LN -> fp32 trunk.
// ---------------------------------------------------------------------------
__global__ __launch_bounds__(96)
void patch_embed_k(const float* __restrict__ img, const float* __restrict__ w,
                   const float* __restrict__ bias, const float* __restrict__ g,
                   const float* __restrict__ bta, float* __restrict__ out)
{
    int t = blockIdx.x;
    int b = t / 3136, l = t % 3136;
    int oy = l / 56, ox = l % 56;
    int c = threadIdx.x;
    float acc = bias[c];
    const float* wp = w + c * 48;
    for (int ic = 0; ic < 3; ++ic)
        for (int ky = 0; ky < 4; ++ky) {
            const float* ip = img + (((size_t)b * 3 + ic) * 224 + oy * 4 + ky) * 224 + ox * 4;
            const float* wr = wp + (ic * 4 + ky) * 4;
#pragma unroll
            for (int kx = 0; kx < 4; ++kx) acc += ip[kx] * wr[kx];
        }
    __shared__ float sh[96];
    sh[c] = acc;
    __syncthreads();
    float s = 0.f, s2 = 0.f;
    for (int i = 0; i < 96; ++i) { float v = sh[i]; s += v; s2 += v * v; }
    float mu = s / 96.0f;
    float rstd = rsqrtf(s2 / 96.0f - mu * mu + 1e-5f);
    out[((size_t)b * 3136 + l) * 96 + c] = (acc - mu) * rstd * g[c] + bta[c];
}

// ---------------------------------------------------------------------------
// Patch merge: 2x2 gather -> 4C concat + LN -> bf16.
// ---------------------------------------------------------------------------
__global__ __launch_bounds__(128)
void merge_ln_k(const float* __restrict__ x, const float* __restrict__ g,
                const float* __restrict__ bta, u16* __restrict__ out,
                int B, int H, int C)
{
    int H2 = H / 2, L2 = H2 * H2;
    int row = blockIdx.x * 4 + (threadIdx.x >> 5);
    if (row >= B * L2) return;
    int lane = threadIdx.x & 31;
    int b = row / L2, l = row % L2;
    int i = l / H2, j = l % H2;
    int C4 = 4 * C;
    const float* xb = x + (size_t)b * H * H * C;
    float s = 0.f, s2 = 0.f;
    for (int cc = lane; cc < C4; cc += 32) {
        int part = cc / C, ch = cc - part * C;
        int dy = part & 1, dx = part >> 1;
        float v = xb[((2 * i + dy) * H + (2 * j + dx)) * C + ch];
        s += v; s2 += v * v;
    }
#pragma unroll
    for (int d = 1; d < 32; d <<= 1) { s += __shfl_xor(s, d, 32); s2 += __shfl_xor(s2, d, 32); }
    float mu = s / C4;
    float rstd = rsqrtf(s2 / C4 - mu * mu + 1e-5f);
    u16* orow = out + (size_t)row * C4;
    for (int cc = lane; cc < C4; cc += 32) {
        int part = cc / C, ch = cc - part * C;
        int dy = part & 1, dx = part >> 1;
        float v = xb[((2 * i + dy) * H + (2 * j + dx)) * C + ch];
        orow[cc] = bfbits((v - mu) * rstd * g[cc] + bta[cc]);
    }
}

// ---------------------------------------------------------------------------
// (B,L,C) fp32 -> (B,C,L) fp32 stage outputs.
// ---------------------------------------------------------------------------
__global__ __launch_bounds__(256)
void transpose_out_k(const float* __restrict__ x, float* __restrict__ out,
                     int B, int L, int C)
{
    size_t idx = (size_t)blockIdx.x * 256 + threadIdx.x;
    size_t tot = (size_t)B * L * C;
    if (idx >= tot) return;
    int c = (int)(idx % C);
    size_t r = idx / C;
    int l = (int)(r % L);
    int b = (int)(r / L);
    out[((size_t)b * C + c) * L + l] = x[idx];
}

// ---------------------------------------------------------------------------
// fp32 -> bf16 weight conversion (once per launch).
// ---------------------------------------------------------------------------
__global__ __launch_bounds__(256)
void cvt_f32_bf16_k(const float* __restrict__ x, u16* __restrict__ o, long long n)
{
    long long i = (long long)blockIdx.x * 256 + threadIdx.x;
    if (i < n) o[i] = bfbits(x[i]);
}

// ---------------------------------------------------------------------------
// Host orchestration. Input flattening = setup_inputs() insertion order.
// ---------------------------------------------------------------------------
static const int kDEPTH[4] = {2, 2, 6, 2};
static const int kDIM[4]   = {96, 192, 384, 768};
static const int kHEAD[4]  = {3, 6, 12, 24};
static const int kRES[4]   = {56, 28, 14, 7};

extern "C" void kernel_launch(void* const* d_in, const int* in_sizes, int n_in,
                              void* d_out, int out_size, void* d_ws, size_t ws_size,
                              hipStream_t stream)
{
    int idx = 0;
    auto nxt = [&]() { return (const float*)d_in[idx++]; };

    const float* image = nxt();
    const float* pe_w  = nxt();
    const float* pe_b  = nxt();
    const float* pe_ng = nxt();
    const float* pe_nb = nxt();

    const float* blk[4][6][13];
    const float* down[3][3];
    for (int s = 0; s < 4; ++s) {
        for (int bi = 0; bi < kDEPTH[s]; ++bi)
            for (int t = 0; t < 13; ++t) blk[s][bi][t] = nxt();
        if (s < 3)
            for (int t = 0; t < 3; ++t) down[s][t] = nxt();
    }

    // workspace layout (bytes): x fp32 | h bf16 | ao bf16 | big bf16 | weights bf16
    char* wsb = (char*)d_ws;
    const size_t X0 = (size_t)32 * 3136 * 96;   // max B*L*C elements
    float* x   = (float*)wsb;                   // 4*X0 B
    u16*   h   = (u16*)(wsb + 4 * X0);          // 2*X0 B
    u16*   ao  = (u16*)(wsb + 6 * X0);          // 2*X0 B
    u16*   big = (u16*)(wsb + 8 * X0);          // 8*X0 B (qkv / mlp / merge)
    u16*   wtab = (u16*)(wsb + 16 * X0);        // ~55 MB of bf16 weights

    // ---- convert all GEMM weights to bf16 once ----------------------------
    size_t woff = 0;
    auto cvt = [&](const float* src, size_t n) -> const u16* {
        u16* dst = wtab + woff;
        woff += (n + 7) & ~(size_t)7;           // keep 16B alignment
        cvt_f32_bf16_k<<<(unsigned)((n + 255) / 256), 256, 0, stream>>>(src, dst, (long long)n);
        return dst;
    };
    const u16 *wq[4][6], *wpj[4][6], *wf1[4][6], *wf2[4][6], *wrd[3];
    for (int s = 0; s < 4; ++s) {
        const size_t C = kDIM[s];
        for (int bi = 0; bi < kDEPTH[s]; ++bi) {
            wq[s][bi]  = cvt(blk[s][bi][2],  3 * C * C);
            wpj[s][bi] = cvt(blk[s][bi][5],  C * C);
            wf1[s][bi] = cvt(blk[s][bi][9],  4 * C * C);
            wf2[s][bi] = cvt(blk[s][bi][11], 4 * C * C);
        }
        if (s < 3) wrd[s] = cvt(down[s][2], (2 * C) * (4 * C));
    }

    patch_embed_k<<<32 * 3136, 96, 0, stream>>>(image, pe_w, pe_b, pe_ng, pe_nb, x);

    float* outp = (float*)d_out;
    size_t ooff = 0;

    for (int s = 0; s < 4; ++s) {
        const int C = kDIM[s], HW = kRES[s], L = HW * HW, M = 32 * L, nH = kHEAD[s];
        const int nW = (HW / 7) * (HW / 7);

        for (int bi = 0; bi < kDEPTH[s]; ++bi) {
            const float* const* p = blk[s][bi];
            const int shift = (HW > 7 && (bi & 1)) ? 3 : 0;

            // --- attention branch ---
            layernorm_k<<<(M + 3) / 4, 128, 0, stream>>>(x, p[0], p[1], h, M, C);
            gemm_bf16_wmma<<<dim3((3 * C + 127) / 128, (M + 63) / 64), 128, 0, stream>>>(
                h, wq[s][bi], p[3], nullptr, big, M, 3 * C, C, 0, 1);
            win_attn<<<32 * nW * nH, 32, 0, stream>>>(big, p[4], ao, HW, nH, C, shift);
            gemm_bf16_wmma<<<dim3((C + 127) / 128, (M + 63) / 64), 128, 0, stream>>>(
                ao, wpj[s][bi], p[6], x, x, M, C, C, 0, 0);

            // --- MLP branch ---
            layernorm_k<<<(M + 3) / 4, 128, 0, stream>>>(x, p[7], p[8], h, M, C);
            gemm_bf16_wmma<<<dim3((4 * C + 127) / 128, (M + 63) / 64), 128, 0, stream>>>(
                h, wf1[s][bi], p[10], nullptr, big, M, 4 * C, C, 1, 1);
            gemm_bf16_wmma<<<dim3((C + 127) / 128, (M + 63) / 64), 128, 0, stream>>>(
                big, wf2[s][bi], p[12], x, x, M, C, 4 * C, 0, 0);
        }

        if (s < 3) {
            const int M2 = 32 * (L / 4);
            merge_ln_k<<<(M2 + 3) / 4, 128, 0, stream>>>(x, down[s][0], down[s][1], big, 32, HW, C);
            gemm_bf16_wmma<<<dim3((2 * C + 127) / 128, (M2 + 63) / 64), 128, 0, stream>>>(
                big, wrd[s], nullptr, nullptr, x, M2, 2 * C, 4 * C, 0, 0);
            const size_t n = (size_t)M2 * 2 * C;
            transpose_out_k<<<(unsigned)((n + 255) / 256), 256, 0, stream>>>(x, outp + ooff, 32, L / 4, 2 * C);
            ooff += n;
        } else {
            const size_t n = (size_t)M * C;
            transpose_out_k<<<(unsigned)((n + 255) / 256), 256, 0, stream>>>(x, outp + ooff, 32, L, C);
            ooff += n;
        }
    }
}
